// PowerLawLayer_678604832902
// MI455X (gfx1250) — matrix operand order repaired and checked by reading, not compiled, not run on hardware
//
#include <hip/hip_runtime.h>

typedef __attribute__((ext_vector_type(16))) _Float16 v16h;
typedef __attribute__((ext_vector_type(8)))  float    v8f;

#define BATCH   64
#define TSTEPS  1024
#define INPUT   256
#define HIDDEN  512
#define NGATE   1536                   // 3*HIDDEN
#define ROWS    (BATCH * TSTEPS)       // 65536
#define EPSC    1.0e-3f
#define PVAL    0.2f
#define LOG2E   1.4426950408889634f

// Recurrence kernel geometry: 64 blocks x 64 threads (2 waves).
#define RBLOCKS   64
#define RTHREADS  64
#define GRP_BLKS  16                   // blocks per batch group
#define LDS_ROWS  96                   // 2 waves * 3 gates * 16 rows
#define LDS_BYTES (LDS_ROWS * HIDDEN * 2)   // 96 KB

// Phase-1 register blocking: N tiles per wave
#define PG_NB 4

// ---------------------------------------------------------------------------
// Fast CDNA5 transcendentals (v_exp_f32 = 2^x, v_log_f32 = log2, v_rcp_f32)
// ---------------------------------------------------------------------------
__device__ __forceinline__ float fast_sigmoid(float x) {
  return __builtin_amdgcn_rcpf(1.0f + __builtin_amdgcn_exp2f(-LOG2E * x));
}
__device__ __forceinline__ float fast_tanh(float x) {
  return 1.0f -
         2.0f * __builtin_amdgcn_rcpf(
                    1.0f + __builtin_amdgcn_exp2f(2.0f * LOG2E * x));
}
__device__ __forceinline__ float fast_pow_p(float x) {  // x^PVAL, x > 0
  return __builtin_amdgcn_exp2f(PVAL * __builtin_amdgcn_logf(x));
}

// ---------------------------------------------------------------------------
// WMMA fragment loaders (wave32, 16x16x32 f16 -> f32).  Layouts used here (contiguous k, not ISA table order):
//  A (16x32 MxK): lane m=lane&15, hb=lane>>4; a[j]=A[m][k0+16hb+j]
//  B (32x16 KxN) with B[k][n]=W[n][k]: lane n=lane&15; b[j]=W[n][k0+16hb+j]
//  C/D f32: c[i] = C[8*hb + i][n]
// ---------------------------------------------------------------------------
__device__ __forceinline__ v16h load_a_f16(const _Float16* A, long row0, int ld,
                                           int k0, int lane) {
  const int m = lane & 15, hb = lane >> 4;
  const _Float16* p = A + (row0 + m) * (long)ld + (k0 + 16 * hb);
  v16h a;
#pragma unroll
  for (int j = 0; j < 8; ++j) a[j] = p[j];
#pragma unroll
  for (int j = 0; j < 8; ++j) a[8 + j] = p[8 + j];
  return a;
}

__device__ __forceinline__ v16h load_a_f32(const float* A, long row0, int ld,
                                           int k0, int lane) {
  const int m = lane & 15, hb = lane >> 4;
  const float* p = A + (row0 + m) * (long)ld + (k0 + 16 * hb);
  v16h a;
#pragma unroll
  for (int j = 0; j < 8; ++j) a[j] = (_Float16)p[j];
#pragma unroll
  for (int j = 0; j < 8; ++j) a[8 + j] = (_Float16)p[8 + j];
  return a;
}

__device__ __forceinline__ v16h load_b_wT(const _Float16* W, int n0, int ldw,
                                          int k0, int lane) {
  const int n = lane & 15, hb = lane >> 4;
  const _Float16* p = W + (long)(n0 + n) * ldw + (k0 + 16 * hb);
  v16h b;
#pragma unroll
  for (int j = 0; j < 16; ++j) b[j] = p[j];
  return b;
}

// B fragment from LDS (row-major [LDS_ROWS][HIDDEN] f16), rbase = local row
__device__ __forceinline__ v16h load_b_lds(const _Float16* sB, int rbase,
                                           int k0, int lane) {
  const int n = lane & 15, hb = lane >> 4;
  const _Float16* p = sB + (rbase + n) * HIDDEN + (k0 + 16 * hb);
  v16h b;
#pragma unroll
  for (int j = 0; j < 16; ++j) b[j] = p[j];
  return b;
}

__device__ __forceinline__ v8f wmma_f16(v16h a, v16h b, v8f c) {
  return __builtin_amdgcn_wmma_f32_16x16x32_f16(false, a, false, b, (short)0, c,
                                                false, false);
}

// ---------------------------------------------------------------------------
// Kernel 0: weights fp32 -> f16
// ---------------------------------------------------------------------------
__global__ void plw_cvt_weights(const float* __restrict__ Wih,
                                const float* __restrict__ Whh,
                                _Float16* __restrict__ WihH,
                                _Float16* __restrict__ WhhH) {
  const int nih = NGATE * INPUT;
  const int nhh = NGATE * HIDDEN;
  for (int i = blockIdx.x * blockDim.x + threadIdx.x; i < nhh;
       i += gridDim.x * blockDim.x) {
    if (i < nih) WihH[i] = (_Float16)Wih[i];
    WhhH[i] = (_Float16)Whh[i];
  }
}

// ---------------------------------------------------------------------------
// Kernel 1: state init every call (h=0 both buffers, c=0, k=-1, counters=0)
// ---------------------------------------------------------------------------
__global__ void plw_init_state(_Float16* __restrict__ h0,
                               _Float16* __restrict__ h1,
                               float* __restrict__ c, float* __restrict__ k,
                               unsigned* __restrict__ cnt) {
  const int n = BATCH * HIDDEN;
  const int gid = blockIdx.x * blockDim.x + threadIdx.x;
  if (gid < 4) cnt[gid] = 0u;
  for (int i = gid; i < n; i += gridDim.x * blockDim.x) {
    h0[i] = (_Float16)0.0f;
    h1[i] = (_Float16)0.0f;
    c[i] = 0.0f;
    k[i] = -1.0f;
  }
}

// ---------------------------------------------------------------------------
// Kernel 2: pre-gates for ALL timesteps:  pre = X @ Wih^T + bias
// Register-blocked: each wave computes 1 M-tile x PG_NB N-tiles, so each
// A fragment (fp32 X converted to f16 on the fly) feeds PG_NB WMMAs.
// ---------------------------------------------------------------------------
__global__ void plw_pregate_gemm(const float* __restrict__ X,
                                 const _Float16* __restrict__ WihH,
                                 const float* __restrict__ bias,
                                 float* __restrict__ pre) {
  const int NTG = NGATE / (16 * PG_NB);  // 24 N-groups
  const int MT = ROWS / 16;              // 4096
  const int wave = (int)((blockIdx.x * blockDim.x + threadIdx.x) >> 5);
  const int lane = threadIdx.x & 31;
  if (wave >= MT * NTG) return;  // wave-uniform
  const int mt = wave / NTG;
  const int ntg = wave - mt * NTG;
  const long row0 = (long)mt * 16;
  const int n0 = ntg * (16 * PG_NB);
  const int n = lane & 15, hb = lane >> 4;

  v8f acc[PG_NB];
#pragma unroll
  for (int q = 0; q < PG_NB; ++q) {
    const float bv = bias[n0 + 16 * q + n];
#pragma unroll
    for (int i = 0; i < 8; ++i) acc[q][i] = bv;
  }

#pragma unroll
  for (int k0 = 0; k0 < INPUT; k0 += 32) {
    if (k0 + 32 < INPUT)
      __builtin_prefetch(X + (row0 + (lane & 15)) * (long)INPUT + k0 + 32, 0, 1);
    v16h a = load_a_f32(X, row0, INPUT, k0, lane);
#pragma unroll
    for (int q = 0; q < PG_NB; ++q) {
      v16h b = load_b_wT(WihH, n0 + 16 * q, INPUT, k0, lane);
      acc[q] = wmma_f16(a, b, acc[q]);
    }
  }

#pragma unroll
  for (int q = 0; q < PG_NB; ++q) {
    float* pp = pre + (row0 + 8 * hb) * (long)NGATE + n0 + 16 * q + n;
#pragma unroll
    for (int i = 0; i < 8; ++i) pp[(long)i * NGATE] = acc[q][i];
  }
}

// ---------------------------------------------------------------------------
// Kernel 3: persistent recurrence — ALL 1024 steps in one launch.
//  - W_hh tiles for this block staged once into LDS (96 KB), reused 1024x.
//  - h ping-pongs between two global f16 buffers each step.
//  - Per-batch-group software barrier: monotonic atomic counter + s_sleep spin.
// ---------------------------------------------------------------------------
__global__ void __launch_bounds__(RTHREADS)
plw_recurrence(const _Float16* __restrict__ WhhH, const float* __restrict__ pre,
               _Float16* __restrict__ h0, _Float16* __restrict__ h1,
               float* __restrict__ cstate, float* __restrict__ kstate,
               float* __restrict__ out, unsigned* __restrict__ cnt) {
  extern __shared__ __align__(16) char smem_raw[];
  _Float16* sB = (_Float16*)smem_raw;

  const int lane = threadIdx.x & 31;
  const int wv   = threadIdx.x >> 5;        // 0..1
  const int g    = blockIdx.x >> 4;         // 0..3  batch group
  const int cb   = blockIdx.x & 15;         // 0..15 column sub-block
  const int m0   = g * 16;                  // batch rows [m0, m0+16)
  const int jt   = cb * 2 + wv;             // 0..31
  const int j0   = jt * 16;                 // hidden cols [j0, j0+16)
  const int n = lane & 15, hb = lane >> 4;

  // ---- Stage this block's W_hh rows into LDS (96 rows x 512 f16) ----
  const int chunks = LDS_ROWS * HIDDEN / 8;  // 8 halves (16B) per chunk
  for (int cidx = threadIdx.x; cidx < chunks; cidx += blockDim.x) {
    const int hoff = cidx * 8;
    const int row = hoff >> 9;            // / HIDDEN
    const int col = hoff & (HIDDEN - 1);
    const int lwv = row / 48;
    const int rem = row - lwv * 48;
    const int gate = rem >> 4;
    const int nloc = rem & 15;
    const int grow = gate * HIDDEN + (cb * 2 + lwv) * 16 + nloc;
    const uint4 v = *(const uint4*)(WhhH + (long)grow * HIDDEN + col);
    *(uint4*)(sB + row * HIDDEN + col) = v;
  }
  __syncthreads();

  const int rb_r = (wv * 3 + 0) * 16;   // LDS row bases for r/o/g
  const int rb_o = (wv * 3 + 1) * 16;
  const int rb_g = (wv * 3 + 2) * 16;

  float* hfp = out + (long)ROWS * HIDDEN;   // h_f [B,H]
  float* cfp = hfp + BATCH * HIDDEN;        // c_f
  float* kfp = cfp + BATCH * HIDDEN;        // k_f

  for (int t = 0; t < TSTEPS; ++t) {
    const _Float16* hr = (t & 1) ? h1 : h0;
    _Float16* hw       = (t & 1) ? h0 : h1;

    // Seed accumulators with pre-gate tiles: pre[(b*T + t)*NGATE + col]
    v8f ar, ao, ag;
#pragma unroll
    for (int i = 0; i < 8; ++i) {
      const long b = m0 + 8 * hb + i;
      const long pr = (b * TSTEPS + t) * (long)NGATE;
      ar[i] = pre[pr + j0 + n];
      ao[i] = pre[pr + HIDDEN + j0 + n];
      ag[i] = pre[pr + 2 * HIDDEN + j0 + n];
    }
    if (t + 1 < TSTEPS) {  // warm caches for next step's pre-gate tile
      const long b = m0 + 8 * hb;
      __builtin_prefetch(pre + (b * TSTEPS + t + 1) * (long)NGATE + j0 + n, 0, 1);
    }

#pragma unroll 4
    for (int k0 = 0; k0 < HIDDEN; k0 += 32) {
      v16h a  = load_a_f16(hr, m0, HIDDEN, k0, lane);
      v16h br = load_b_lds(sB, rb_r, k0, lane);
      v16h bo = load_b_lds(sB, rb_o, k0, lane);
      v16h bg = load_b_lds(sB, rb_g, k0, lane);
      ar = wmma_f16(a, br, ar);
      ao = wmma_f16(a, bo, ao);
      ag = wmma_f16(a, bg, ag);
    }

    const float tf = (float)t;
#pragma unroll
    for (int i = 0; i < 8; ++i) {
      const long b = m0 + 8 * hb + i;
      const int j = j0 + n;
      const long sidx = b * HIDDEN + j;

      const float r  = fast_sigmoid(ar[i]);
      const float o  = fast_sigmoid(ao[i]);
      const float gg = fast_tanh(ag[i]);

      const float kold = kstate[sidx];
      const float cold = cstate[sidx];
      const float knew = r * (tf - EPSC) + (1.0f - r) * kold;
      const float x = (tf - knew + EPSC) *
                      __builtin_amdgcn_rcpf(tf - knew + 1.0f);  // in (0,1)
      const float f = fast_pow_p(x);                            // x^0.2
      const float cnew = f * cold + (1.0f - f) * gg;
      const float h = o * fast_tanh(cnew);

      cstate[sidx] = cnew;
      kstate[sidx] = knew;
      hw[sidx] = (_Float16)h;
      out[(b * TSTEPS + t) * (long)HIDDEN + j] = h;
      if (t == TSTEPS - 1) { hfp[sidx] = h; cfp[sidx] = cnew; kfp[sidx] = knew; }
    }

    // ---- group barrier: all GRP_BLKS blocks of group g finished step t ----
    __syncthreads();
    __threadfence();  // release this block's h stores to agent scope
    if (threadIdx.x == 0) {
      __hip_atomic_fetch_add(&cnt[g], 1u, __ATOMIC_ACQ_REL,
                             __HIP_MEMORY_SCOPE_AGENT);
      const unsigned target = (unsigned)GRP_BLKS * (unsigned)(t + 1);
      while (__hip_atomic_load(&cnt[g], __ATOMIC_ACQUIRE,
                               __HIP_MEMORY_SCOPE_AGENT) < target) {
        __builtin_amdgcn_s_sleep(1);
      }
    }
    __syncthreads();
    __threadfence();  // acquire: invalidate before reading peers' h
  }
}

// ---------------------------------------------------------------------------
// Host launcher
// ---------------------------------------------------------------------------
extern "C" void kernel_launch(void* const* d_in, const int* in_sizes, int n_in,
                              void* d_out, int out_size, void* d_ws,
                              size_t ws_size, hipStream_t stream) {
  const float* X    = (const float*)d_in[0];   // [64,1024,256]
  const float* Wih  = (const float*)d_in[1];   // [1536,256]
  const float* Whh  = (const float*)d_in[2];   // [1536,512]
  const float* bias = (const float*)d_in[3];   // [1536]
  float* out = (float*)d_out;

  char* ws = (char*)d_ws;
  size_t off = 0;
  auto carve = [&](size_t bytes) {
    char* p = ws + off;
    off = (off + bytes + 255) & ~(size_t)255;
    return p;
  };
  _Float16* WihH = (_Float16*)carve((size_t)NGATE * INPUT * 2);
  _Float16* WhhH = (_Float16*)carve((size_t)NGATE * HIDDEN * 2);
  _Float16* h0   = (_Float16*)carve((size_t)BATCH * HIDDEN * 2);
  _Float16* h1   = (_Float16*)carve((size_t)BATCH * HIDDEN * 2);
  float*    cst  = (float*)carve((size_t)BATCH * HIDDEN * 4);
  float*    kst  = (float*)carve((size_t)BATCH * HIDDEN * 4);
  unsigned* cnt  = (unsigned*)carve(4 * sizeof(unsigned));
  float*    pre  = (float*)carve((size_t)ROWS * NGATE * 4);   // 384 MB
  (void)ws_size; (void)in_sizes; (void)n_in; (void)out_size;

  // 0) weights -> f16
  plw_cvt_weights<<<1024, 256, 0, stream>>>(Wih, Whh, WihH, WhhH);

  // 1) state + barrier-counter init
  plw_init_state<<<128, 256, 0, stream>>>(h0, h1, cst, kst, cnt);

  // 2) pre-gates: 4096 M-tiles x 24 N-groups = 98304 waves, 8 waves/block
  plw_pregate_gemm<<<12288, 256, 0, stream>>>(X, WihH, bias, pre);

  // 3) persistent recurrence: one launch, 64 blocks x 2 waves, 96 KB LDS each
  plw_recurrence<<<RBLOCKS, RTHREADS, LDS_BYTES, stream>>>(
      WhhH, pre, h0, h1, cst, kst, out, cnt);
}
